// TransformerVAE_85152021610759
// MI455X (gfx1250) — matrix operand order repaired and checked by reading, not compiled, not run on hardware
//
#include <hip/hip_runtime.h>
#include <hip/hip_bf16.h>

// ---------------------------------------------------------------------------
// CDNA5 (gfx1250) TransformerVAE forward: bf16 WMMA GEMMs with TDM-staged
// weight panels (tensor_load_to_lds + TENSORcnt double buffering) + flash
// attention.
// ---------------------------------------------------------------------------

#define D_MODEL 256
#define NHEAD   8
#define DH      32
#define NLAYERS 6
#define DFF     1024
#define ZDIM    4
#define BATCH   8
#define SEQ     1024
#define MTOT    (BATCH * SEQ)    // 8192
#define ZHDIM   (D_MODEL + ZDIM) // 260

typedef __attribute__((ext_vector_type(16))) __bf16 v16bf;
typedef __attribute__((ext_vector_type(8)))  float  v8f;
typedef __attribute__((ext_vector_type(4)))  unsigned int u32x4;
typedef __attribute__((ext_vector_type(8)))  unsigned int u32x8;

union U16x8 { unsigned int u[8]; v16bf v; };

__device__ inline v8f zero8() {
    v8f v;
#pragma unroll
    for (int i = 0; i < 8; ++i) v[i] = 0.f;
    return v;
}

__device__ inline unsigned short f2bf(float f) {
    unsigned int u = __float_as_uint(f);
    u += 0x7FFFu + ((u >> 16) & 1u);   // round-to-nearest-even
    return (unsigned short)(u >> 16);
}

__device__ inline float redmax16(float v) {
#pragma unroll
    for (int m = 1; m < 16; m <<= 1) v = fmaxf(v, __shfl_xor(v, m, 32));
    return v;
}
__device__ inline float redsum16(float v) {
#pragma unroll
    for (int m = 1; m < 16; m <<= 1) v += __shfl_xor(v, m, 32);
    return v;
}
__device__ inline float redsum32(float v) {
#pragma unroll
    for (int m = 16; m >= 1; m >>= 1) v += __shfl_xor(v, m, 32);
    return v;
}
__device__ inline float softplusf(float x) {
    if (x > 20.f) return x;
    return log1pf(expf(x));
}

// ---------------------------------------------------------------------------
// TDM: async 2D tile load Global -> LDS (D# groups 0/1; groups 2/3 NULL).
//   tile:  tile_k (dim0, contiguous) x rows (dim1)
//   tensor: tensor_k x tensor_rows, row stride = stride_elems (bf16 units)
// ---------------------------------------------------------------------------
__device__ inline void tdm_load_2d_bf16(const unsigned short* gaddr,
                                        unsigned lds_off,
                                        unsigned tile_k, unsigned rows,
                                        unsigned tensor_k, unsigned tensor_rows,
                                        unsigned stride_elems) {
    unsigned long long ga = (unsigned long long)(__SIZE_TYPE__)gaddr;
    u32x4 g0;
    g0[0] = 1u;                                   // count=1 (valid), user mode
    g0[1] = lds_off;                              // lds_addr (bytes)
    g0[2] = (unsigned)(ga & 0xFFFFFFFFu);         // global_addr[31:0]
    g0[3] = (unsigned)((ga >> 32) & 0x01FFFFFFu)  // global_addr[56:32]
            | (2u << 30);                         // type = 2 ("image")
    u32x8 g1;
    g1[0] = (1u << 16);                           // data_size=1 -> 2 bytes
    g1[1] = (tensor_k & 0xFFFFu) << 16;           // tensor_dim0[15:0] @ bits 63:48
    g1[2] = (tensor_k >> 16) | ((tensor_rows & 0xFFFFu) << 16); // dim0 hi | dim1 lo
    g1[3] = (tensor_rows >> 16) | (tile_k << 16); // dim1 hi | tile_dim0
    g1[4] = rows;                                 // tile_dim1 (tile_dim2 = 0)
    g1[5] = stride_elems;                         // tensor_dim0_stride[31:0]
    g1[6] = 0;                                    // stride hi | dim1_stride lo
    g1[7] = 0;
    asm volatile("tensor_load_to_lds %0, %1" :: "s"(g0), "s"(g1) : "memory");
}

// ---------------------------------------------------------------------------
// fp32 -> bf16 weight conversion
// ---------------------------------------------------------------------------
__global__ void cvt_bf16_kernel(const float* __restrict__ in,
                                unsigned short* __restrict__ out, int n) {
    for (int i = blockIdx.x * blockDim.x + threadIdx.x; i < n;
         i += gridDim.x * blockDim.x)
        out[i] = f2bf(in[i]);
}

// ---------------------------------------------------------------------------
// conv1d(k=3,'SAME') embedding + sinusoidal posenc  ->  h (fp32 + bf16)
// ---------------------------------------------------------------------------
__device__ inline float getx(const float* x, int b, int p, int lag) {
    if (p < 0 || p >= SEQ) return 0.f;
    if (lag) { if (p == 0) return 0.f; return x[b * SEQ + p - 1]; }
    return x[b * SEQ + p];
}

__global__ void embed_kernel(const float* __restrict__ x,
                             const float* __restrict__ cw,
                             const float* __restrict__ cb,
                             float* __restrict__ hf,
                             unsigned short* __restrict__ hb, int lag) {
    int row = blockIdx.x;            // b*SEQ + l
    int b = row >> 10, l = row & (SEQ - 1);
    int d = threadIdx.x;
    float acc = cb[d];
#pragma unroll
    for (int k = 0; k < 3; ++k)
        acc += getx(x, b, l + k - 1, lag) * cw[k * D_MODEL + d];
    int i2 = d >> 1;
    float freq = expf(-(float)(2 * i2) * (9.210340371976184f / (float)D_MODEL));
    float ang = (float)l * freq;
    acc += (d & 1) ? cosf(ang) : sinf(ang);
    size_t idx = (size_t)row * D_MODEL + d;
    hf[idx] = acc;
    hb[idx] = f2bf(acc);
}

// ---------------------------------------------------------------------------
// GEMM:  C[M,N] = A[M,K] (bf16) * W[N,K]^T (bf16) + bias, fp32 accumulate.
// block = 256 threads = 8 waves (2 x 4); wave tile = 16 rows x 64 cols.
// Weight panel (256 N-rows x 64 K) staged by the Tensor Data Mover into
// double-buffered LDS; wave 0 drives TENSORcnt, workgroup barriers publish.
// grid = (N/256, M/32)
// ---------------------------------------------------------------------------
#define TDK 64   // K elements per TDM chunk

__global__ void __launch_bounds__(256)
gemm_bf16_wmma(const unsigned short* __restrict__ A,
               const unsigned short* __restrict__ W,
               const float* __restrict__ bias,
               float* __restrict__ Cf,
               unsigned short* __restrict__ Cb,
               int N, int K, int relu) {
    __shared__ unsigned short wtile[2][256 * TDK];   // 2 x 32 KB
    const int lane = threadIdx.x & 31;
    const int wave = threadIdx.x >> 5;
    const int half = lane >> 4;
    const int ln   = lane & 15;
    const int wr = wave >> 2;        // 0..1
    const int wc = wave & 3;         // 0..3
    const int m0 = blockIdx.y * 32 + wr * 16;
    const int nb = blockIdx.x * 256 + wc * 64;
    const int nbase = blockIdx.x * 256;              // panel base row of W

    v8f acc0 = zero8(), acc1 = zero8(), acc2 = zero8(), acc3 = zero8();

    const unsigned short* arow = A + (size_t)(m0 + ln) * K;
    const int nchunks = K / TDK;
    const unsigned lds0 = (unsigned)(__SIZE_TYPE__)(&wtile[0][0]);
    const unsigned lds1 = (unsigned)(__SIZE_TYPE__)(&wtile[1][0]);

    if (wave == 0) {
        tdm_load_2d_bf16(W + (size_t)nbase * K, lds0, TDK, 256,
                         (unsigned)K, 256u, (unsigned)K);
    }
    for (int c = 0; c < nchunks; ++c) {
        if (wave == 0) {
            if (c + 1 < nchunks) {
                tdm_load_2d_bf16(W + (size_t)nbase * K + (c + 1) * TDK,
                                 ((c + 1) & 1) ? lds1 : lds0, TDK, 256,
                                 (unsigned)K, 256u, (unsigned)K);
                __builtin_amdgcn_s_wait_tensorcnt(1);
            } else {
                __builtin_amdgcn_s_wait_tensorcnt(0);
            }
        }
        __syncthreads();                     // publish chunk c to all waves
        const unsigned short* wb = &wtile[c & 1][0];
        // prefetch next A chunk while consuming this one
        __builtin_prefetch(arow + (c + 1) * TDK, 0, 3);
#pragma unroll
        for (int kk = 0; kk < 2; ++kk) {
            const int k0 = c * TDK + kk * 32;
            U16x8 a;
#pragma unroll
            for (int p = 0; p < 8; ++p) {
                int k = 2 * p + 16 * half;
                a.u[p] = *(const unsigned int*)(arow + k0 + k);
            }
#pragma unroll
            for (int t = 0; t < 4; ++t) {
                const unsigned short* wrow =
                    wb + ((wc * 64 + t * 16 + ln) * TDK + kk * 32 + 16 * half);
                U16x8 bfr;
#pragma unroll
                for (int p = 0; p < 8; ++p)
                    bfr.u[p] = *(const unsigned int*)(wrow + 2 * p);
                v8f cprev = (t == 0) ? acc0 : (t == 1) ? acc1 : (t == 2) ? acc2 : acc3;
                v8f cnew = __builtin_amdgcn_wmma_f32_16x16x32_bf16(
                    false, a.v, false, bfr.v, (short)0, cprev, false, false);
                if (t == 0) acc0 = cnew; else if (t == 1) acc1 = cnew;
                else if (t == 2) acc2 = cnew; else acc3 = cnew;
            }
        }
        __syncthreads();                     // reads done before next overwrite
    }
    // epilogue
#pragma unroll
    for (int t = 0; t < 4; ++t) {
        v8f c = (t == 0) ? acc0 : (t == 1) ? acc1 : (t == 2) ? acc2 : acc3;
        int col = nb + t * 16 + ln;
        float bs = bias ? bias[col] : 0.f;
#pragma unroll
        for (int r = 0; r < 8; ++r) {
            int m = m0 + r + 8 * half;
            float v = c[r] + bs;
            if (relu) v = fmaxf(v, 0.f);
            size_t idx = (size_t)m * N + col;
            if (Cf) Cf[idx] = v;
            if (Cb) Cb[idx] = f2bf(v);
        }
    }
}

// ---------------------------------------------------------------------------
// Causal flash attention, one wave per (batch, head, 16-query tile).
// qkv: [MTOT, 768] bf16 (q | k | v).  out: [MTOT, 256] bf16.
// grid = (64, 8, 8), block = 32.
// ---------------------------------------------------------------------------
__global__ void __launch_bounds__(32)
attn_kernel(const unsigned short* __restrict__ qkv,
            unsigned short* __restrict__ out) {
    const int lane = threadIdx.x & 31;
    const int half = lane >> 4;
    const int ln   = lane & 15;
    const int qt = blockIdx.x, hd = blockIdx.y, b = blockIdx.z;
    __shared__ unsigned short ldsP[16 * 32];

    // Q fragment (A: 16x32, dh == WMMA-K)
    U16x8 aq;
    {
        size_t tok = (size_t)b * SEQ + qt * 16 + ln;
        const unsigned short* qp = qkv + tok * 768 + hd * DH;
#pragma unroll
        for (int p = 0; p < 8; ++p) {
            int k = 2 * p + 16 * half;
            aq.u[p] = *(const unsigned int*)(qp + k);
        }
    }
    v8f o0 = zero8(), o1 = zero8();
    float rm[8], rs[8];
#pragma unroll
    for (int r = 0; r < 8; ++r) { rm[r] = -3.0e38f; rs[r] = 0.f; }
    const float sc = 0.17677669529663687f;   // 1/sqrt(32)
    const int nsteps = qt / 2 + 1;

    for (int st = 0; st < nsteps; ++st) {
        const int kb = st * 32;
        // prefetch next K-panel row for this lane
        __builtin_prefetch(qkv + ((size_t)b * SEQ + kb + 32 + ln) * 768
                               + D_MODEL + hd * DH, 0, 3);
        v8f s0, s1;
#pragma unroll
        for (int kk = 0; kk < 2; ++kk) {
            size_t ktok = (size_t)b * SEQ + kb + kk * 16 + ln;
            const unsigned short* kp = qkv + ktok * 768 + D_MODEL + hd * DH;
            U16x8 bk;
#pragma unroll
            for (int p = 0; p < 8; ++p) {
                int k = 2 * p + 16 * half;
                bk.u[p] = *(const unsigned int*)(kp + k);
            }
            v8f sres = __builtin_amdgcn_wmma_f32_16x16x32_bf16(
                false, aq.v, false, bk.v, (short)0, zero8(), false, false);
            if (kk == 0) s0 = sres; else s1 = sres;
        }
        __syncthreads();                 // prior-step P reads complete
#pragma unroll
        for (int r = 0; r < 8; ++r) {
            int m = r + 8 * half;
            int qi = qt * 16 + m;
            float x0 = (kb + ln <= qi) ? s0[r] * sc : -3.0e38f;
            float x1 = (kb + 16 + ln <= qi) ? s1[r] * sc : -3.0e38f;
            float mt = redmax16(fmaxf(x0, x1));
            float nm = fmaxf(rm[r], mt);
            float p0 = expf(x0 - nm);
            float p1 = expf(x1 - nm);
            float alpha = expf(rm[r] - nm);
            rs[r] = rs[r] * alpha + redsum16(p0 + p1);
            rm[r] = nm;
            o0[r] *= alpha;
            o1[r] *= alpha;
            ldsP[m * 32 + ln]      = f2bf(p0);
            ldsP[m * 32 + 16 + ln] = f2bf(p1);
        }
        __syncthreads();
        U16x8 ap;
#pragma unroll
        for (int p = 0; p < 8; ++p) {
            int k = 2 * p + 16 * half;
            ap.u[p] = *(const unsigned int*)(&ldsP[ln * 32 + k]);
        }
#pragma unroll
        for (int t = 0; t < 2; ++t) {
            U16x8 bv;
            int d = t * 16 + ln;
#pragma unroll
            for (int p = 0; p < 8; ++p) {
                int k = 2 * p + 16 * half;
                size_t t0 = (size_t)b * SEQ + kb + k;
                unsigned int lo = qkv[t0 * 768 + 512 + hd * DH + d];
                unsigned int hi = qkv[(t0 + 1) * 768 + 512 + hd * DH + d];
                bv.u[p] = lo | (hi << 16);
            }
            if (t == 0)
                o0 = __builtin_amdgcn_wmma_f32_16x16x32_bf16(
                    false, ap.v, false, bv.v, (short)0, o0, false, false);
            else
                o1 = __builtin_amdgcn_wmma_f32_16x16x32_bf16(
                    false, ap.v, false, bv.v, (short)0, o1, false, false);
        }
    }
#pragma unroll
    for (int r = 0; r < 8; ++r) {
        int m = r + 8 * half;
        size_t tok = (size_t)b * SEQ + qt * 16 + m;
        float inv = 1.f / rs[r];
        out[tok * D_MODEL + hd * DH + ln]      = f2bf(o0[r] * inv);
        out[tok * D_MODEL + hd * DH + 16 + ln] = f2bf(o1[r] * inv);
    }
}

// ---------------------------------------------------------------------------
// residual add + LayerNorm  ->  h (fp32 + bf16).  grid = MTOT, block = 256.
// ---------------------------------------------------------------------------
__global__ void __launch_bounds__(256)
add_ln_kernel(const float* __restrict__ res, const float* __restrict__ o,
              const float* __restrict__ g, const float* __restrict__ bta,
              float* __restrict__ hf, unsigned short* __restrict__ hb) {
    __shared__ float red[256];
    int row = blockIdx.x, tid = threadIdx.x;
    size_t idx = (size_t)row * D_MODEL + tid;
    float v = res[idx] + o[idx];
    red[tid] = v; __syncthreads();
    for (int s = 128; s > 0; s >>= 1) {
        if (tid < s) red[tid] += red[tid + s];
        __syncthreads();
    }
    float mu = red[0] / (float)D_MODEL; __syncthreads();
    float d = v - mu;
    red[tid] = d * d; __syncthreads();
    for (int s = 128; s > 0; s >>= 1) {
        if (tid < s) red[tid] += red[tid + s];
        __syncthreads();
    }
    float rstd = rsqrtf(red[0] / (float)D_MODEL + 1e-5f);
    float y = d * rstd * g[tid] + bta[tid];
    hf[idx] = y;
    hb[idx] = f2bf(y);
}

// ---------------------------------------------------------------------------
// Sequential latent scan.  1 block x 256 threads; warp w = batch b.
// ---------------------------------------------------------------------------
__global__ void __launch_bounds__(256)
scan_kernel(const float* __restrict__ h1, const float* __restrict__ eps,
            const float* __restrict__ lw, const float* __restrict__ lb,
            const float* __restrict__ sw, const float* __restrict__ sb,
            float* __restrict__ z, float* __restrict__ zl,
            float* __restrict__ zs) {
    int w = threadIdx.x >> 5, lane = threadIdx.x & 31;
    int b = w;
    __shared__ float sh[BATCH][ZHDIM];
    __shared__ float zn[BATCH][ZDIM];
    if (lane < ZDIM) sh[w][lane] = 0.f;
    for (int t = 0; t < SEQ; ++t) {
        for (int c = lane; c < D_MODEL; c += 32)
            sh[w][ZDIM + c] = h1[((size_t)b * SEQ + t) * D_MODEL + c];
        __syncthreads();
#pragma unroll
        for (int j = 0; j < ZDIM; ++j) {
            float a = 0.f, s = 0.f;
            for (int c = lane; c < ZHDIM; c += 32) {
                float v = sh[w][c];
                a += lw[j * ZHDIM + c] * v;
                s += sw[j * ZHDIM + c] * v;
            }
            a = redsum32(a) + lb[j];
            s = softplusf(redsum32(s) + sb[j]);
            if (lane == 0) {
                float e = eps[((size_t)t * BATCH + b) * ZDIM + j];
                float znew = a + s * e;
                size_t idx = ((size_t)b * SEQ + t) * ZDIM + j;
                zl[idx] = a; zs[idx] = s; z[idx] = znew;
                zn[w][j] = znew;
            }
        }
        __syncthreads();
        if (lane < ZDIM) sh[w][lane] = zn[w][lane];
        __syncthreads();
    }
}

// ---------------------------------------------------------------------------
// Heads + losses.  warp per token row; grid = MTOT/8, block = 256.
// ---------------------------------------------------------------------------
__global__ void __launch_bounds__(256)
heads_kernel(const float* __restrict__ x, const float* __restrict__ h2,
             const float* __restrict__ z, const float* __restrict__ zl,
             const float* __restrict__ zs,
             const float* __restrict__ xlw, const float* __restrict__ xlb,
             const float* __restrict__ xsw, const float* __restrict__ xsb,
             const float* __restrict__ zlw, const float* __restrict__ zlb,
             const float* __restrict__ zsw, const float* __restrict__ zsb,
             float* __restrict__ out) {
    int w = threadIdx.x >> 5, lane = threadIdx.x & 31;
    int row = blockIdx.x * 8 + w;
    __shared__ float sh[8][D_MODEL];
    for (int c = lane; c < D_MODEL; c += 32)
        sh[w][c] = h2[(size_t)row * D_MODEL + c];
    __syncthreads();
    int l = row & (SEQ - 1);
    float zr[ZDIM], zg[ZDIM];
#pragma unroll
    for (int j = 0; j < ZDIM; ++j) {
        zr[j] = z[(size_t)row * ZDIM + j];
        zg[j] = (l > 0) ? z[(size_t)(row - 1) * ZDIM + j] : 0.f;
    }
    float pl = 0.f, ps = 0.f;
    for (int c = lane; c < ZHDIM; c += 32) {
        float v = (c < ZDIM) ? zr[c] : sh[w][c - ZDIM];
        pl += xlw[c] * v;
        ps += xsw[c] * v;
    }
    float yl  = redsum32(pl) + xlb[0];
    float ysc = softplusf(redsum32(ps) + xsb[0]);
    float zpl[ZDIM], zps[ZDIM];
#pragma unroll
    for (int j = 0; j < ZDIM; ++j) {
        float a = 0.f, bm = 0.f;
        for (int c = lane; c < ZHDIM; c += 32) {
            float v = (c < D_MODEL) ? sh[w][c] : zg[c - D_MODEL];
            a  += zlw[j * ZHDIM + c] * v;
            bm += zsw[j * ZHDIM + c] * v;
        }
        zpl[j] = redsum32(a) + zlb[j];
        zps[j] = softplusf(redsum32(bm) + zsb[j]);
    }
    if (lane == 0) {
        out[row] = yl;
        float xv = x[row];
        float t = (xv - yl) / ysc;
        float rec = logf(ysc) + 0.9189385332046727f + 0.5f * t * t;
        float kl = 0.f;
#pragma unroll
        for (int j = 0; j < ZDIM; ++j) {
            float s  = zs[(size_t)row * ZDIM + j];
            float mu = zl[(size_t)row * ZDIM + j];
            float dd = mu - zpl[j];
            kl += logf(zps[j] / s) + (s * s + dd * dd) / (2.f * zps[j] * zps[j]) - 0.5f;
        }
        atomicAdd(&out[MTOT], rec);
        atomicAdd(&out[MTOT + 1], kl);
    }
}

__global__ void zero_losses_kernel(float* out) {
    out[MTOT] = 0.f;
    out[MTOT + 1] = 0.f;
}

// ---------------------------------------------------------------------------
// launch
// ---------------------------------------------------------------------------
extern "C" void kernel_launch(void* const* d_in, const int* in_sizes, int n_in,
                              void* d_out, int out_size, void* d_ws, size_t ws_size,
                              hipStream_t stream) {
    const float* x       = (const float*)d_in[0];
    const float* eps     = (const float*)d_in[1];
    const float* conv_w  = (const float*)d_in[2];
    const float* conv_b  = (const float*)d_in[3];
    const float* win     = (const float*)d_in[4];
    const float* bin     = (const float*)d_in[5];
    const float* wo      = (const float*)d_in[6];
    const float* bo      = (const float*)d_in[7];
    const float* ln1g    = (const float*)d_in[8];
    const float* ln1b    = (const float*)d_in[9];
    const float* w1      = (const float*)d_in[10];
    const float* b1      = (const float*)d_in[11];
    const float* w2      = (const float*)d_in[12];
    const float* b2      = (const float*)d_in[13];
    const float* ln2g    = (const float*)d_in[14];
    const float* ln2b    = (const float*)d_in[15];
    const float* phi_lw  = (const float*)d_in[16];
    const float* phi_lb  = (const float*)d_in[17];
    const float* phi_sw  = (const float*)d_in[18];
    const float* phi_sb  = (const float*)d_in[19];
    const float* thx_lw  = (const float*)d_in[20];
    const float* thx_lb  = (const float*)d_in[21];
    const float* thx_sw  = (const float*)d_in[22];
    const float* thx_sb  = (const float*)d_in[23];
    const float* thz_lw  = (const float*)d_in[24];
    const float* thz_lb  = (const float*)d_in[25];
    const float* thz_sw  = (const float*)d_in[26];
    const float* thz_sb  = (const float*)d_in[27];
    float* out = (float*)d_out;

    // workspace carve-up
    char* p = (char*)d_ws;
    auto take = [&](size_t bytes) {
        char* r = p;
        p += (bytes + 255) & ~(size_t)255;
        return r;
    };
    const int WIN_N = 3 * D_MODEL * D_MODEL;      // 768*256 per layer
    const int WO_N  = D_MODEL * D_MODEL;
    const int W1_N  = DFF * D_MODEL;
    const int W2_N  = D_MODEL * DFF;
    unsigned short* win_bf = (unsigned short*)take((size_t)NLAYERS * WIN_N * 2);
    unsigned short* wo_bf  = (unsigned short*)take((size_t)NLAYERS * WO_N * 2);
    unsigned short* w1_bf  = (unsigned short*)take((size_t)NLAYERS * W1_N * 2);
    unsigned short* w2_bf  = (unsigned short*)take((size_t)NLAYERS * W2_N * 2);
    float*          h_f32  = (float*)take((size_t)MTOT * D_MODEL * 4);
    unsigned short* h_bf   = (unsigned short*)take((size_t)MTOT * D_MODEL * 2);
    unsigned short* qkv_bf = (unsigned short*)take((size_t)MTOT * 3 * D_MODEL * 2);
    unsigned short* att_bf = (unsigned short*)take((size_t)MTOT * D_MODEL * 2);
    float*          o_f32  = (float*)take((size_t)MTOT * D_MODEL * 4);
    unsigned short* ff_bf  = (unsigned short*)take((size_t)MTOT * DFF * 2);
    float* zbuf  = (float*)take((size_t)MTOT * ZDIM * 4);
    float* zlbuf = (float*)take((size_t)MTOT * ZDIM * 4);
    float* zsbuf = (float*)take((size_t)MTOT * ZDIM * 4);

    // convert weights to bf16
    cvt_bf16_kernel<<<512, 256, 0, stream>>>(win, win_bf, NLAYERS * WIN_N);
    cvt_bf16_kernel<<<512, 256, 0, stream>>>(wo,  wo_bf,  NLAYERS * WO_N);
    cvt_bf16_kernel<<<512, 256, 0, stream>>>(w1,  w1_bf,  NLAYERS * W1_N);
    cvt_bf16_kernel<<<512, 256, 0, stream>>>(w2,  w2_bf,  NLAYERS * W2_N);
    zero_losses_kernel<<<1, 1, 0, stream>>>(out);

    for (int pass = 0; pass < 2; ++pass) {
        embed_kernel<<<MTOT, 256, 0, stream>>>(x, conv_w, conv_b, h_f32, h_bf, pass);
        for (int i = 0; i < NLAYERS; ++i) {
            // QKV: [8192,256] x [768,256]^T
            gemm_bf16_wmma<<<dim3(3, MTOT / 32), 256, 0, stream>>>(
                h_bf, win_bf + (size_t)i * WIN_N, bin + i * 3 * D_MODEL,
                nullptr, qkv_bf, 3 * D_MODEL, D_MODEL, 0);
            attn_kernel<<<dim3(SEQ / 16, NHEAD, BATCH), 32, 0, stream>>>(qkv_bf, att_bf);
            // output projection
            gemm_bf16_wmma<<<dim3(1, MTOT / 32), 256, 0, stream>>>(
                att_bf, wo_bf + (size_t)i * WO_N, bo + i * D_MODEL,
                o_f32, nullptr, D_MODEL, D_MODEL, 0);
            add_ln_kernel<<<MTOT, 256, 0, stream>>>(
                h_f32, o_f32, ln1g + i * D_MODEL, ln1b + i * D_MODEL, h_f32, h_bf);
            // FF1 (ReLU)
            gemm_bf16_wmma<<<dim3(4, MTOT / 32), 256, 0, stream>>>(
                h_bf, w1_bf + (size_t)i * W1_N, b1 + i * DFF,
                nullptr, ff_bf, DFF, D_MODEL, 1);
            // FF2
            gemm_bf16_wmma<<<dim3(1, MTOT / 32), 256, 0, stream>>>(
                ff_bf, w2_bf + (size_t)i * W2_N, b2 + i * D_MODEL,
                o_f32, nullptr, D_MODEL, DFF, 0);
            add_ln_kernel<<<MTOT, 256, 0, stream>>>(
                h_f32, o_f32, ln2g + i * D_MODEL, ln2b + i * D_MODEL, h_f32, h_bf);
        }
        if (pass == 0) {
            scan_kernel<<<1, 256, 0, stream>>>(h_f32, eps, phi_lw, phi_lb,
                                               phi_sw, phi_sb, zbuf, zlbuf, zsbuf);
        }
    }
    heads_kernel<<<MTOT / 8, 256, 0, stream>>>(
        x, h_f32, zbuf, zlbuf, zsbuf,
        thx_lw, thx_lb, thx_sw, thx_sb,
        thz_lw, thz_lb, thz_sw, thz_sb, out);
    (void)in_sizes; (void)n_in; (void)out_size; (void)ws_size;
}
